// RandLANet_18296560681033
// MI455X (gfx1250) — matrix operand order repaired and checked by reading, not compiled, not run on hardware
//
#include <hip/hip_runtime.h>
#include <math.h>

// ---------------------------------------------------------------------------
// Types for CDNA5 WMMA (wave32, 16x16x32 f16 -> f32 accumulate)
// ---------------------------------------------------------------------------
typedef __attribute__((ext_vector_type(16))) _Float16 v16h;
typedef __attribute__((ext_vector_type(8)))  _Float16 v8h;
typedef __attribute__((ext_vector_type(8)))  float    v8f;

// ---------------------------------------------------------------------------
// Weight prep: Wh[o][c] = f16(W[o][c]) zero-padded to [Opad][Cpad].
// Zero pad in the C (=WMMA K) dimension makes unguarded B loads safe:
// A pad rows are 0, so garbage B values at k >= C contribute nothing.
// ---------------------------------------------------------------------------
__global__ void k_prepW(const float* __restrict__ W, _Float16* __restrict__ Wh,
                        int O, int C, int Opad, int Cpad)
{
    long total  = (long)Opad * Cpad;
    long stride = (long)gridDim.x * blockDim.x;
    for (long t = (long)blockIdx.x * blockDim.x + threadIdx.x; t < total; t += stride) {
        int o = (int)(t / Cpad);
        int c = (int)(t - (long)o * Cpad);
        Wh[t] = (o < O && c < C) ? (_Float16)W[(long)o * C + c] : (_Float16)0.f;
    }
}

// ---------------------------------------------------------------------------
// WMMA GEMM:  Y[b][o][m] = sum_c W[o][c] * X[b][c][m] + bias[o]
//   Wh: [Opad][Cpad] f16, zero padded. X base + b*xBS, Y base + b*yBS.
//   M is a multiple of 16 for every layer in this network.
// Block = 128 threads = 4 waves; each wave computes one 16x16 tile.
// All addresses are (uniform base + 32-bit unsigned offset) so the backend
// can use saddr global loads with cheap 32-bit offset arithmetic.
// ---------------------------------------------------------------------------
__global__ __launch_bounds__(128)
void k_gemm(const _Float16* __restrict__ Wh, const float* __restrict__ Xm,
            const float* __restrict__ bias, float* __restrict__ Ym,
            int O, int C, int Cpad, int M, long xBS, long yBS)
{
    const int b    = blockIdx.z;
    const float* X = Xm + (long)b * xBS;
    float*       Y = Ym + (long)b * yBS;

    const int wave = threadIdx.x >> 5;
    const int lane = threadIdx.x & 31;
    const int m0   = (blockIdx.x * 4 + wave) * 16;
    if (m0 >= M) return;                       // wave-uniform: EXEC stays all-1s
    const int o0   = blockIdx.y * 16;

    const int row = lane & 15;
    const int hi  = lane >> 4;

    const unsigned Mu    = (unsigned)M;
    const unsigned mcol  = (unsigned)(m0 + row);
    const unsigned aoff  = (unsigned)(o0 + row) * (unsigned)Cpad + (unsigned)(hi * 16);
    const unsigned bbase = (unsigned)(hi * 16) * Mu + mcol;   // K-half base

    v8f acc = {};
    const int Clim = (C / 32) * 32;

    int c0 = 0;
#pragma unroll 2
    for (; c0 < Clim; c0 += 32) {
        v8h a0 = *(const v8h*)(Wh + aoff + (unsigned)c0);
        v8h a1 = *(const v8h*)(Wh + aoff + (unsigned)c0 + 8u);
        v16h af = __builtin_shufflevector(a0, a1, 0, 1, 2, 3, 4, 5, 6, 7,
                                          8, 9, 10, 11, 12, 13, 14, 15);
        const unsigned bo = (unsigned)c0 * Mu + bbase;
        v16h bf;
#pragma unroll
        for (int idx = 0; idx < 16; ++idx)
            bf[idx] = (_Float16)X[bo + (unsigned)idx * Mu];

        // gfx1250 prefetch path (global_prefetch_b8) for the next K-slab of X.
        if (c0 + 32 < Clim)
            __builtin_prefetch(X + bo + 32u * Mu, 0, 1);

        acc = __builtin_amdgcn_wmma_f32_16x16x32_f16(
            false, af, false, bf, (short)0, acc, false, false);
    }

    if (Clim < C) {  // ragged tail: clamp B addresses; A pad zeros mask them
        v8h a0 = *(const v8h*)(Wh + aoff + (unsigned)c0);
        v8h a1 = *(const v8h*)(Wh + aoff + (unsigned)c0 + 8u);
        v16h af = __builtin_shufflevector(a0, a1, 0, 1, 2, 3, 4, 5, 6, 7,
                                          8, 9, 10, 11, 12, 13, 14, 15);
        v16h bf;
#pragma unroll
        for (int idx = 0; idx < 16; ++idx) {
            int c = c0 + hi * 16 + idx;
            if (c > C - 1) c = C - 1;          // address clamp only, no select
            bf[idx] = (_Float16)X[(unsigned)c * Mu + mcol];
        }
        acc = __builtin_amdgcn_wmma_f32_16x16x32_f16(
            false, af, false, bf, (short)0, acc, false, false);
    }

    // D layout: VGPR v holds row M = v + hi*8, col N = lane&15.
#pragma unroll
    for (int v = 0; v < 8; ++v) {
        int oo = o0 + v + hi * 8;
        if (oo < O) Y[(long)oo * M + mcol] = acc[v] + bias[oo];
    }
}

// ---------------------------------------------------------------------------
// BatchNorm statistics, deterministic two-pass reduction.
// Pass 1: grid (S, C); each block reduces a strided slice of [B x M].
// ---------------------------------------------------------------------------
__global__ __launch_bounds__(256)
void k_bn_part(const float* __restrict__ Y, float* __restrict__ part,
               long M, long bStride, int B, int S)
{
    __shared__ float s1[256], s2[256];
    const int s = blockIdx.x;
    const int c = blockIdx.y;
    const float* base = Y + (long)c * M;
    float sum = 0.f, sq = 0.f;
    const long step = (long)S * blockDim.x;
    for (int b = 0; b < B; ++b) {
        const float* p = base + (long)b * bStride;
        for (long m = (long)s * blockDim.x + threadIdx.x; m < M; m += step) {
            float v = p[m];
            sum += v;
            sq  += v * v;
        }
    }
    s1[threadIdx.x] = sum; s2[threadIdx.x] = sq;
    __syncthreads();
    for (int t = 128; t > 0; t >>= 1) {
        if ((int)threadIdx.x < t) {
            s1[threadIdx.x] += s1[threadIdx.x + t];
            s2[threadIdx.x] += s2[threadIdx.x + t];
        }
        __syncthreads();
    }
    if (threadIdx.x == 0) {
        part[((long)c * S + s) * 2 + 0] = s1[0];
        part[((long)c * S + s) * 2 + 1] = s2[0];
    }
}

// Pass 2: grid (C); stats[2c] = mean, stats[2c+1] = rsqrt(var + 1e-6)
__global__ void k_bn_fin(const float* __restrict__ part, float* __restrict__ stats,
                         int S, float invN)
{
    const int c = blockIdx.x;
    if (threadIdx.x == 0) {
        float x = 0.f, y = 0.f;
        for (int i = 0; i < S; ++i) {
            x += part[((long)c * S + i) * 2 + 0];
            y += part[((long)c * S + i) * 2 + 1];
        }
        float mean = x * invN;
        float var  = y * invN - mean * mean;
        stats[2 * c]     = mean;
        stats[2 * c + 1] = rsqrtf(var + 1e-6f);
    }
}

// BN apply (+ optional leaky relu, slope < 0 => none), in place.
// grid: (ceil(M/256), C, B)
__global__ void k_bn_apply(float* __restrict__ Y, const float* __restrict__ stats,
                           const float* __restrict__ gg, const float* __restrict__ beta,
                           long M, long bStride, float slope)
{
    const int  c = blockIdx.y;
    const long b = blockIdx.z;
    const long m = (long)blockIdx.x * blockDim.x + threadIdx.x;
    if (m >= M) return;
    float* p = Y + b * bStride + (long)c * M + m;
    float v = *p;
    v = (v - stats[2 * c]) * stats[2 * c + 1] * gg[c] + beta[c];
    if (slope >= 0.f) v = (v >= 0.f) ? v : slope * v;
    *p = v;
}

// ---------------------------------------------------------------------------
// Relative point features: out channels = [dist, rel(3), ext(3), nbrc(3)]
// coords [B][N][3], nbr [B][N][16], out [B][10][N*16]; grid (ceil(NK/256), B)
// ---------------------------------------------------------------------------
__global__ void k_relfeat(const float* __restrict__ coords, const int* __restrict__ nbr,
                          float* __restrict__ out, int N)
{
    const long NK = (long)N * 16;
    const long b  = blockIdx.y;
    const long r  = (long)blockIdx.x * blockDim.x + threadIdx.x;
    if (r >= NK) return;
    const int n = (int)(r >> 4);
    const int j = nbr[b * NK + r];
    const float* cb = coords + b * (long)N * 3;
    float ex = cb[(long)n * 3 + 0], ey = cb[(long)n * 3 + 1], ez = cb[(long)n * 3 + 2];
    float nx = cb[(long)j * 3 + 0], ny = cb[(long)j * 3 + 1], nz = cb[(long)j * 3 + 2];
    float rx = ex - nx, ry = ey - ny, rz = ez - nz;
    float d  = sqrtf(rx * rx + ry * ry + rz * rz);
    float* ob = out + b * 10 * NK + r;
    ob[0 * NK] = d;
    ob[1 * NK] = rx; ob[2 * NK] = ry; ob[3 * NK] = rz;
    ob[4 * NK] = ex; ob[5 * NK] = ey; ob[6 * NK] = ez;
    ob[7 * NK] = nx; ob[8 * NK] = ny; ob[9 * NK] = nz;
}

// Neighbor gather: dst[b][c][p] = src[b][c][nbr[b][p]]; grid (ceil(NK/256), C, B)
__global__ void k_gather(const float* __restrict__ src, const int* __restrict__ nbr,
                         float* __restrict__ dst, int srcN, long NK,
                         long srcBS, long dstBS)
{
    const int  c = blockIdx.y;
    const long b = blockIdx.z;
    const long p = (long)blockIdx.x * blockDim.x + threadIdx.x;
    if (p >= NK) return;
    const int j = nbr[b * NK + p];
    dst[b * dstBS + (long)c * NK + p] = src[b * srcBS + (long)c * srcN + j];
}

// Attention pool: softmax over K=16 of S, dotted with X -> F[b][c][n]
// grid (ceil(N/256), C, B)
__global__ void k_attpool(const float* __restrict__ S, const float* __restrict__ X,
                          float* __restrict__ F, int N,
                          long sBS, long xBS, long fBS)
{
    const int  c = blockIdx.y;
    const long b = blockIdx.z;
    const long n = (long)blockIdx.x * blockDim.x + threadIdx.x;
    if (n >= N) return;
    const float* sp = S + b * sBS + ((long)c * N + n) * 16;
    const float* xp = X + b * xBS + ((long)c * N + n) * 16;
    float mx = -3.4e38f;
#pragma unroll
    for (int k = 0; k < 16; ++k) mx = fmaxf(mx, sp[k]);
    float den = 0.f, num = 0.f;
#pragma unroll
    for (int k = 0; k < 16; ++k) {
        float e = __expf(sp[k] - mx);
        den += e;
        num += e * xp[k];
    }
    F[b * fBS + (long)c * N + n] = num / den;
}

// Copy channel block: dst[b][c][m] = src[b][c][m]; grid (ceil(M/256), C, B)
__global__ void k_copy(const float* __restrict__ src, float* __restrict__ dst,
                       long M, long srcBS, long dstBS)
{
    const int  c = blockIdx.y;
    const long b = blockIdx.z;
    const long m = (long)blockIdx.x * blockDim.x + threadIdx.x;
    if (m >= M) return;
    dst[b * dstBS + (long)c * M + m] = src[b * srcBS + (long)c * M + m];
}

// Elementwise add + leaky relu
__global__ void k_add_lrelu(const float* __restrict__ a, const float* __restrict__ bb,
                            float* __restrict__ o, long total, float slope)
{
    long stride = (long)gridDim.x * blockDim.x;
    for (long t = (long)blockIdx.x * blockDim.x + threadIdx.x; t < total; t += stride) {
        float v = a[t] + bb[t];
        o[t] = (v >= 0.f) ? v : slope * v;
    }
}

// Max-pool subsample: dst[b][c][n2] = max_k src[b][c][sub[b][n2][k]]
// grid (ceil(N2/256), C, B), K = 16
__global__ void k_maxpool(const float* __restrict__ src, const int* __restrict__ sub,
                          float* __restrict__ dst, int N2, int srcN,
                          long srcBS, long dstBS)
{
    const int  c = blockIdx.y;
    const long b = blockIdx.z;
    const long n = (long)blockIdx.x * blockDim.x + threadIdx.x;
    if (n >= N2) return;
    const int* ip = sub + b * (long)N2 * 16 + n * 16;
    const float* sp = src + b * srcBS + (long)c * srcN;
    float mx = -3.4e38f;
#pragma unroll
    for (int k = 0; k < 16; ++k) mx = fmaxf(mx, sp[ip[k]]);
    dst[b * dstBS + (long)c * N2 + n] = mx;
}

// Nearest interpolation (upsample): dst[b][c][n] = src[b][c][idx[b][n]]
// grid (ceil(N/256), C, B)
__global__ void k_interp(const float* __restrict__ src, const int* __restrict__ idx,
                         float* __restrict__ dst, int N, int srcN,
                         long srcBS, long dstBS)
{
    const int  c = blockIdx.y;
    const long b = blockIdx.z;
    const long n = (long)blockIdx.x * blockDim.x + threadIdx.x;
    if (n >= N) return;
    const int j = idx[b * (long)N + n];
    dst[b * dstBS + (long)c * N + n] = src[b * srcBS + (long)c * srcN + j];
}

// fc0: out[b][f][n] = sum_c feats[b][n][c]*W[f][c] + bias[f]  (C=6, F=8)
__global__ void k_fc0(const float* __restrict__ f, const float* __restrict__ W,
                      const float* __restrict__ bias, float* __restrict__ out,
                      int N, int B)
{
    long total  = (long)B * N;
    long stride = (long)gridDim.x * blockDim.x;
    for (long t = (long)blockIdx.x * blockDim.x + threadIdx.x; t < total; t += stride) {
        long b = t / N;
        int  n = (int)(t - b * N);
        const float* fp = f + ((long)b * N + n) * 6;
        float x0 = fp[0], x1 = fp[1], x2 = fp[2], x3 = fp[3], x4 = fp[4], x5 = fp[5];
        float* ob = out + b * 8L * N;
#pragma unroll
        for (int o = 0; o < 8; ++o) {
            const float* wr = W + o * 6;
            ob[(long)o * N + n] = bias[o] + wr[0] * x0 + wr[1] * x1 + wr[2] * x2 +
                                  wr[3] * x3 + wr[4] * x4 + wr[5] * x5;
        }
    }
}

// Output transpose: o[b][n][cls] = x[b][cls][n]
__global__ void k_out(const float* __restrict__ x, float* __restrict__ o,
                      int N, int CLS, int B)
{
    long total  = (long)B * N * CLS;
    long stride = (long)gridDim.x * blockDim.x;
    for (long t = (long)blockIdx.x * blockDim.x + threadIdx.x; t < total; t += stride) {
        long b = t / ((long)N * CLS);
        long r = t - b * (long)N * CLS;
        int  n = (int)(r / CLS);
        int  c = (int)(r - (long)n * CLS);
        o[t] = x[b * (long)CLS * N + (long)c * N + n];
    }
}

// ---------------------------------------------------------------------------
// Host-side orchestration
// ---------------------------------------------------------------------------
namespace {

struct Smlp { const float *W, *b, *beta, *g; };
struct Pool { Smlp mlp; const float *sW, *sb; };
struct Cur  { void* const* din; int i;
              const float* f() { return (const float*)din[i++]; } };

Smlp takeS(Cur& c, bool bn = true) {
    Smlp s; s.W = c.f(); s.b = c.f();
    if (bn) { s.beta = c.f(); s.g = c.f(); } else { s.beta = nullptr; s.g = nullptr; }
    return s;
}
Pool takeP(Cur& c) { Pool p; p.mlp = takeS(c); p.sW = c.f(); p.sb = c.f(); return p; }

inline int nb(long total) {
    long b = (total + 255) / 256;
    return (int)(b > 262144 ? 262144 : (b < 1 ? 1 : b));
}
inline dim3 g3(long x, int c, int b) {
    return dim3((unsigned)((x + 255) / 256), (unsigned)c, (unsigned)b);
}

// GEMM with weight prep (f32 -> padded f16), stream-ordered slab reuse.
inline void gemm(hipStream_t st, _Float16* WHp, const float* W, const float* X,
                 const float* bias, float* Y, int O, int C, int M,
                 long xBS, long yBS) {
    const int Opad = (O + 15) & ~15;
    const int Cpad = (C + 31) & ~31;
    k_prepW<<<nb((long)Opad * Cpad), 256, 0, st>>>(W, WHp, O, C, Opad, Cpad);
    dim3 g((unsigned)((M / 16 + 3) / 4), (unsigned)(Opad / 16), 2u);
    k_gemm<<<g, 128, 0, st>>>(WHp, X, bias, Y, O, C, Cpad, M, xBS, yBS);
}

inline void bnapply(hipStream_t st, float* PART, float* STATS, float* Y,
                    const float* g_, const float* beta,
                    int C, long M, long bStride, float slope) {
    const int S = 32;
    k_bn_part<<<dim3(S, (unsigned)C), 256, 0, st>>>(Y, PART, M, bStride, 2, S);
    k_bn_fin<<<C, 32, 0, st>>>(PART, STATS, S, 1.f / (float)(2L * M));
    k_bn_apply<<<g3(M, C, 2), 256, 0, st>>>(Y, STATS, g_, beta, M, bStride, slope);
}

} // namespace

extern "C" void kernel_launch(void* const* d_in, const int* in_sizes, int n_in,
                              void* d_out, int out_size, void* d_ws, size_t ws_size,
                              hipStream_t stream)
{
    (void)in_sizes; (void)n_in; (void)out_size; (void)ws_size;
    const hipStream_t st = stream;

    // ---- parameters (pytree sorted-key flatten order), start at index 17 ----
    Cur cur{d_in, 17};
    const float* bn0_beta = cur.f();
    const float* bn0_g    = cur.f();
    Smlp dec[4];
    for (int i = 0; i < 4; ++i) dec[i] = takeS(cur);
    struct EncP { Smlp lse1, lse2, mlp1, mlp2; Pool pool1, pool2; Smlp shortcut; } enc[4];
    for (int i = 0; i < 4; ++i) {
        enc[i].lse1 = takeS(cur); enc[i].lse2 = takeS(cur);
        enc[i].mlp1 = takeS(cur); enc[i].mlp2 = takeS(cur);
        enc[i].pool1 = takeP(cur); enc[i].pool2 = takeP(cur);
        enc[i].shortcut = takeS(cur);
    }
    const float* fc0_W = cur.f();
    const float* fc0_b = cur.f();
    Smlp fc1a = takeS(cur), fc1b = takeS(cur), fc1c = takeS(cur, /*bn=*/false);
    Smlp mid  = takeS(cur);

    // ---- workspace slabs (floats, bump-allocated, 256B aligned) ----
    float* w = (float*)d_ws;
    long off = 0;
    auto take = [&](long n) { float* p = w + off; off += (n + 63) & ~63L; return p; };
    float* STATS = take(2048);
    float* PART  = take(65536);        // BN partial sums (C*S*2 <= 49152)
    float* WHf   = take(140000);       // padded f16 weights (<= 512*512 f16)
    float* FEAT0 = take(2L * 8 * 40960);
    float* X1p   = take(655360);
    float* Ep    = take(655360);
    float* Ap    = take(20971520);     // concat buffer [B][2*d2][N*K]
    float* Bbp   = take(20971520);     // relraw / attention scores / scratch
    float* Dp    = take(1310720);      // pooled features
    float* Fp    = take(1310720);      // pool2 mlp out
    float* Gp    = take(2621440);      // mlp2 out / fe temp / decoder feat
    float* Hp    = take(2621440);      // shortcut out / fc1 temp
    float* FE0p  = take(2621440);      // enc_list[0]
    float* FSp[4];
    FSp[0] = take(655360); FSp[1] = take(655360);
    FSp[2] = take(327680); FSp[3] = take(163840);
    float* M1p   = take(163840);       // mid out
    float* FCp   = take(5242880);      // fc1[0] out
    _Float16* WHp = (_Float16*)WHf;

    const int Ns[5]    = {40960, 10240, 2560, 640, 160};
    const int douts[4] = {16, 64, 128, 256};
    const int Kn = 16;

    // ---- fc0 + bn0 + lrelu(0.2) ----
    k_fc0<<<nb(2L * 40960), 256, 0, st>>>((const float*)d_in[0], fc0_W, fc0_b,
                                          FEAT0, 40960, 2);
    bnapply(st, PART, STATS, FEAT0, bn0_g, bn0_beta, 8, 40960, 8L * 40960, 0.2f);

    // shared_mlp = GEMM + BN(+lrelu)
    auto smlp = [&](const Smlp& p, const float* X, float* Y, int O, int C,
                    long M, long xBS, long yBS, float slope) {
        gemm(st, WHp, p.W, X, p.b, Y, O, C, (int)M, xBS, yBS);
        bnapply(st, PART, STATS, Y, p.g, p.beta, O, M, yBS, slope);
    };

    // ---- encoder ----
    const float* feat = FEAT0;
    int cin = 8;
    for (int i = 0; i < 4; ++i) {
        const int Ni = Ns[i], Ni1 = Ns[i + 1], dO = douts[i], d2 = dO / 2;
        const long NK = (long)Ni * Kn;
        const float* coords = (const float*)d_in[1 + 4 * i];
        const int*   nbr    = (const int*)d_in[2 + 4 * i];
        const int*   sub    = (const int*)d_in[3 + 4 * i];
        const EncP&  P      = enc[i];
        const long aBS = 2L * d2 * NK;   // = dO * NK

        // x1 = mlp1(feat)
        smlp(P.mlp1, feat, X1p, d2, cin, Ni, (long)cin * Ni, (long)d2 * Ni, 0.2f);
        // relraw -> Bb  [B][10][NK]
        k_relfeat<<<g3(NK, 2, 1).x == 0 ? dim3(1, 2) : dim3((unsigned)((NK + 255) / 256), 2),
                    256, 0, st>>>(coords, nbr, Bbp, Ni);
        // relf = lse1(relraw) -> A channels [d2, 2*d2)
        smlp(P.lse1, Bbp, Ap + (long)d2 * NK, d2, 10, NK, 10L * NK, aBS, 0.2f);
        // nbrF = gather(x1) -> A channels [0, d2)
        k_gather<<<g3(NK, d2, 2), 256, 0, st>>>(X1p, nbr, Ap, Ni, NK,
                                                (long)d2 * Ni, aBS);
        // S1 = score_W @ concat1 + score_b -> Bb (relraw consumed)
        gemm(st, WHp, P.pool1.sW, Ap, P.pool1.sb, Bbp, dO, dO, (int)NK, aBS, aBS);
        // attention pool -> D [B][dO][Ni]
        k_attpool<<<g3(Ni, dO, 2), 256, 0, st>>>(Bbp, Ap, Dp, Ni, aBS, aBS,
                                                 (long)dO * Ni);
        // x2 = pool1.mlp(D)
        smlp(P.pool1.mlp, Dp, Ep, d2, dO, Ni, (long)dO * Ni, (long)d2 * Ni, 0.2f);
        // relf2 = lse2(relf) -> Bb
        smlp(P.lse2, Ap + (long)d2 * NK, Bbp, d2, d2, NK, aBS, (long)d2 * NK, 0.2f);
        // concat2: copy relf2 into A[d2..), gather(x2) into A[0..d2)
        k_copy<<<g3(NK, d2, 2), 256, 0, st>>>(Bbp, Ap + (long)d2 * NK, NK,
                                              (long)d2 * NK, aBS);
        k_gather<<<g3(NK, d2, 2), 256, 0, st>>>(Ep, nbr, Ap, Ni, NK,
                                                (long)d2 * Ni, aBS);
        // S2 + pool2 -> D, x3 = pool2.mlp(D) -> F
        gemm(st, WHp, P.pool2.sW, Ap, P.pool2.sb, Bbp, dO, dO, (int)NK, aBS, aBS);
        k_attpool<<<g3(Ni, dO, 2), 256, 0, st>>>(Bbp, Ap, Dp, Ni, aBS, aBS,
                                                 (long)dO * Ni);
        smlp(P.pool2.mlp, Dp, Fp, dO, dO, Ni, (long)dO * Ni, (long)dO * Ni, 0.2f);
        // out = lrelu( bn(mlp2(x3)) + bn(shortcut(feat)), 0.01 )
        smlp(P.mlp2,     Fp,   Gp, 2 * dO, dO,  Ni, (long)dO * Ni,  2L * dO * Ni, -1.f);
        smlp(P.shortcut, feat, Hp, 2 * dO, cin, Ni, (long)cin * Ni, 2L * dO * Ni, -1.f);
        float* feOut = (i == 0) ? FE0p : Gp;
        k_add_lrelu<<<nb(2L * 2 * dO * Ni), 256, 0, st>>>(Gp, Hp, feOut,
                                                          2L * 2 * dO * Ni, 0.01f);
        // fs = max-pool subsample -> persistent slab
        k_maxpool<<<g3(Ni1, 2 * dO, 2), 256, 0, st>>>(feOut, sub, FSp[i], Ni1, Ni,
                                                      2L * dO * Ni, 2L * dO * Ni1);
        feat = FSp[i];
        cin  = 2 * dO;
    }

    // ---- mid ----
    smlp(mid, FSp[3], M1p, 512, 512, 160, 512L * 160, 512L * 160, 0.2f);

    // ---- decoder ----
    const float* encSrc[4] = {FSp[2], FSp[1], FSp[0], FE0p};
    const int    coArr[4]  = {256, 128, 32, 32};
    const float* dfeat = M1p;
    int fC = 512, fN = 160;
    for (int i = 0; i < 4; ++i) {
        const int Nd = Ns[3 - i];
        const int co = coArr[i];
        const int Ct = co + fC;
        const int* idx = (const int*)d_in[4 + 4 * (3 - i)];
        k_copy<<<g3(Nd, co, 2), 256, 0, st>>>(encSrc[i], Ap, Nd,
                                              (long)co * Nd, (long)Ct * Nd);
        k_interp<<<g3(Nd, fC, 2), 256, 0, st>>>(dfeat, idx, Ap + (long)co * Nd,
                                                Nd, fN, (long)fC * fN, (long)Ct * Nd);
        smlp(dec[i], Ap, Gp, co, Ct, Nd, (long)Ct * Nd, (long)co * Nd, 0.2f);
        dfeat = Gp; fC = co; fN = Nd;
    }

    // ---- fc1 head ----
    smlp(fc1a, Gp,  FCp, 64, 32, 40960, 32L * 40960, 64L * 40960, 0.2f);
    smlp(fc1b, FCp, Hp,  32, 64, 40960, 64L * 40960, 32L * 40960, 0.2f);
    gemm(st, WHp, fc1c.W, Hp, fc1c.b, Bbp, 13, 32, 40960, 32L * 40960, 13L * 40960);

    // ---- output transpose (B, N, 13) ----
    k_out<<<nb(2L * 40960 * 13), 256, 0, st>>>(Bbp, (float*)d_out, 40960, 13, 2);
}
